// _HandHistoryEncoder_24584392802976
// MI455X (gfx1250) — compile-verified, matrix operand order checked by reading, not run on hardware
//
#include <hip/hip_runtime.h>
#include <hip/hip_bf16.h>

// ---------------------------------------------------------------------------
// Masked GRU encoder, fused: embeddings + folded input projection + recurrence
// One wave (32 lanes) owns a 16-row batch tile across all T=128 steps.
// All matmuls via v_wmma_f32_16x16x32_f16 (f16 inputs, f32 accumulate).
// ---------------------------------------------------------------------------

typedef _Float16 half_t;
typedef _Float16 v16h __attribute__((ext_vector_type(16)));
typedef float    v8f  __attribute__((ext_vector_type(8)));
typedef unsigned int u32;

#define T_STEPS 128
#define HIDN    64
#define G3      192          // 3*HID gate columns
#define WAVES_PER_BLK 2
#define ROWS_PER_WAVE 16
#define WHH_PITCH 33         // u32 pitch (66 halves) for W_hh rows  (64 used)
#define WC_PITCH  17         // u32 pitch (34 halves) for Wc rows    (32 used)
#define H_PITCH   33         // u32 pitch for h f16 rows             (64 used)
#define F_PITCH   17         // u32 pitch for feats f16 rows         (32 used)

union FragU { v16h h; u32 u[8]; };

__device__ inline v8f wmma_f16(v16h a, v16h b, v8f c) {
  // D = A(16x32 f16) * B(32x16 f16) + C(16x16 f32)
  return __builtin_amdgcn_wmma_f32_16x16x32_f16(
      /*neg_a=*/false, a, /*neg_b=*/false, b,
      /*c_mod=*/(short)0, c, /*reuse_a=*/false, /*reuse_b=*/false);
}

// Assemble a 16x32 f16 operand fragment from LDS.
// Documented 16-bit A layout (B mirrored, lane = N):
//   lane half 0 (lanes 0-15):  V0-3 -> K=0,1..6,7    V4-7 -> K=16..23
//   lane half 1 (lanes 16-31): V0-3 -> K=8..15       V4-7 -> K=24..31
__device__ inline v16h load_frag(const u32* base, int row, int pitchU32,
                                 int kbaseHalf, int lane) {
  const int khalf = lane >> 4;
  const u32* p = base + row * pitchU32 + (kbaseHalf >> 1) + khalf * 4;
  FragU f;
#pragma unroll
  for (int i = 0; i < 4; i++) f.u[i]     = p[i];
#pragma unroll
  for (int i = 0; i < 4; i++) f.u[4 + i] = p[8 + i];
  return f.h;
}

__device__ inline float fast_sigmoid(float x) {
  float e = __builtin_amdgcn_exp2f(-1.44269504f * x);
  return __builtin_amdgcn_rcpf(1.0f + e);
}
__device__ inline float fast_tanh(float x) {
  x = fminf(15.0f, fmaxf(-15.0f, x));
  float e = __builtin_amdgcn_exp2f(2.8853900817779268f * x);
  return (e - 1.0f) * __builtin_amdgcn_rcpf(e + 1.0f);
}

__global__ __launch_bounds__(WAVES_PER_BLK * 32)
void gru_hand_history_kernel(
    const int* __restrict__ actor_ids, const int* __restrict__ action_ids,
    const int* __restrict__ street_ids, const float* __restrict__ bet_fraction,
    const int* __restrict__ valid_mask,
    const float* __restrict__ E_actor, const float* __restrict__ E_action,
    const float* __restrict__ E_street,
    const float* __restrict__ W_proj, const float* __restrict__ b_proj,
    const float* __restrict__ W_ih, const float* __restrict__ W_hh,
    const float* __restrict__ b_ih, const float* __restrict__ b_hh,
    float* __restrict__ out)
{
  // ---- shared weights (per block) -----------------------------------------
  __shared__ u32   sWhh[G3 * WHH_PITCH];   // f16 [192][66]
  __shared__ u32   sWc [G3 * WC_PITCH];    // f16 [192][34]  (Wc = W_ih @ W_proj)
  __shared__ float sBc [G3];               // W_ih@b_proj + b_ih
  __shared__ float sBhh[G3];
  __shared__ float sEa[56], sEac[32], sEs[20];
  // ---- per-wave scratch ---------------------------------------------------
  __shared__ u32 sH16 [WAVES_PER_BLK][ROWS_PER_WAVE * H_PITCH]; // h as f16
  __shared__ u32 sFeat[WAVES_PER_BLK][ROWS_PER_WAVE * F_PITCH]; // feats as f16
  __shared__ int sMask[WAVES_PER_BLK][ROWS_PER_WAVE];

  const int tid   = threadIdx.x;
  const int wave  = tid >> 5;
  const int lane  = tid & 31;
  const int lrow  = lane & 15;   // A: M row / B: N col owned by this lane
  const int khalf = lane >> 4;

  half_t* hWhh = (half_t*)sWhh;
  half_t* hWc  = (half_t*)sWc;

  // ---- weight prep: fold W_proj/b_proj into W_ih, convert to f16 ----------
  for (int i = tid; i < 56; i += blockDim.x) sEa[i]  = E_actor[i];
  for (int i = tid; i < 32; i += blockDim.x) sEac[i] = E_action[i];
  for (int i = tid; i < 20; i += blockDim.x) sEs[i]  = E_street[i];

  for (int g = tid; g < G3; g += blockDim.x) {
    float bacc = b_ih[g];
    for (int k = 0; k < 32; k++) bacc += W_ih[g * 32 + k] * b_proj[k];
    sBc[g]  = bacc;
    sBhh[g] = b_hh[g];
    for (int c = 0; c < 2 * WC_PITCH; c++) {           // 21 live cols, rest 0
      float v = 0.0f;
      if (c < 21)
        for (int k = 0; k < 32; k++) v += W_ih[g * 32 + k] * W_proj[k * 21 + c];
      hWc[g * (2 * WC_PITCH) + c] = (half_t)v;
    }
    for (int k = 0; k < 2 * WHH_PITCH; k++)
      hWhh[g * (2 * WHH_PITCH) + k] = (half_t)((k < HIDN) ? W_hh[g * HIDN + k] : 0.0f);
  }
  for (int w = 0; w < WAVES_PER_BLK; w++) {
    for (int i = tid; i < ROWS_PER_WAVE * H_PITCH; i += blockDim.x) sH16[w][i] = 0u;
    for (int i = tid; i < ROWS_PER_WAVE * F_PITCH; i += blockDim.x) sFeat[w][i] = 0u;
  }
  __syncthreads();

  const int rowBase = blockIdx.x * (WAVES_PER_BLK * ROWS_PER_WAVE)
                    + wave * ROWS_PER_WAVE;
  half_t* hFeatW = (half_t*)sFeat[wave];
  half_t* hH16W  = (half_t*)sH16[wave];

  // hidden state + gates live in registers: element (tile jj, vgpr v) maps to
  // M = v + 8*khalf, N(col) = jj*16 + lrow  (documented C/D layout)
  float h_reg[4][8], r_reg[4][8], z_reg[4][8];
#pragma unroll
  for (int a = 0; a < 4; a++)
#pragma unroll
    for (int b = 0; b < 8; b++) { h_reg[a][b] = 0.0f; r_reg[a][b] = 0.0f; z_reg[a][b] = 0.0f; }

  for (int t = 0; t < T_STEPS; t++) {
    // ---- embed + concat -> f16 feats tile [16 x 21 (pad 32)] --------------
    if (lane < 16) {
      const int row = rowBase + lane;
      const int idx = row * T_STEPS + t;
      const int a  = actor_ids[idx];
      const int ac = action_ids[idx];
      const int st = street_ids[idx];
      const float bet = bet_fraction[idx];
      sMask[wave][lane] = valid_mask[idx];
      half_t* fr = hFeatW + lane * (2 * F_PITCH);
#pragma unroll
      for (int c = 0; c < 8; c++) fr[c]      = (half_t)sEa[a * 8 + c];
#pragma unroll
      for (int c = 0; c < 8; c++) fr[8 + c]  = (half_t)sEac[ac * 8 + c];
#pragma unroll
      for (int c = 0; c < 4; c++) fr[16 + c] = (half_t)sEs[st * 4 + c];
      fr[20] = (half_t)bet;
    }
    __syncthreads();

    const v16h fA  = load_frag(sFeat[wave], lrow, F_PITCH, 0,  lane);
    const v16h hA0 = load_frag(sH16[wave],  lrow, H_PITCH, 0,  lane);
    const v16h hA1 = load_frag(sH16[wave],  lrow, H_PITCH, 32, lane);

    // ---- r (tiles 0-3) and z (tiles 4-7) gates ----------------------------
#pragma unroll
    for (int j = 0; j < 8; j++) {
      const int wrow = j * 16 + lrow;             // weight row = B-operand col
      v8f acc = {};
      acc = wmma_f16(hA0, load_frag(sWhh, wrow, WHH_PITCH, 0,  lane), acc);
      acc = wmma_f16(hA1, load_frag(sWhh, wrow, WHH_PITCH, 32, lane), acc);
      acc = wmma_f16(fA,  load_frag(sWc,  wrow, WC_PITCH,  0,  lane), acc);
      const float bias = sBc[j * 16 + lrow] + sBhh[j * 16 + lrow];
#pragma unroll
      for (int v = 0; v < 8; v++) {
        const float g = fast_sigmoid(acc[v] + bias);
        if (j < 4) r_reg[j][v] = g; else z_reg[j - 4][v] = g;
      }
    }

    // ---- n gate (tiles 8-11): n = tanh(xn + bc + r*(hn + bhh)); update h --
#pragma unroll
    for (int jj = 0; jj < 4; jj++) {
      const int j = 8 + jj;
      const int wrow = j * 16 + lrow;
      v8f accH = {};
      accH = wmma_f16(hA0, load_frag(sWhh, wrow, WHH_PITCH, 0,  lane), accH);
      accH = wmma_f16(hA1, load_frag(sWhh, wrow, WHH_PITCH, 32, lane), accH);
      v8f accX = {};
      accX = wmma_f16(fA,  load_frag(sWc,  wrow, WC_PITCH,  0,  lane), accX);
      const float bX = sBc[j * 16 + lrow];
      const float bH = sBhh[j * 16 + lrow];
#pragma unroll
      for (int v = 0; v < 8; v++) {
        const int   m    = v + 8 * khalf;
        const float nval = fast_tanh(accX[v] + bX + r_reg[jj][v] * (accH[v] + bH));
        const float z    = z_reg[jj][v];
        const float hold = h_reg[jj][v];
        const float hnew = (1.0f - z) * nval + z * hold;
        const float hsel = sMask[wave][m] ? hnew : hold;   // prefix-valid mask
        h_reg[jj][v] = hsel;
        hH16W[m * (2 * H_PITCH) + jj * 16 + lrow] = (half_t)hsel;
      }
    }
    __syncthreads();   // h f16 tile consistent before next step's A-fragments
  }

  // ---- write final hidden state [16 x 64] f32 -----------------------------
#pragma unroll
  for (int jj = 0; jj < 4; jj++)
#pragma unroll
    for (int v = 0; v < 8; v++) {
      const int m   = v + 8 * khalf;
      const int col = jj * 16 + lrow;
      out[(size_t)(rowBase + m) * HIDN + col] = h_reg[jj][v];
    }
}

extern "C" void kernel_launch(void* const* d_in, const int* in_sizes, int n_in,
                              void* d_out, int out_size, void* d_ws, size_t ws_size,
                              hipStream_t stream) {
  const int*   actor_ids    = (const int*)  d_in[0];
  const int*   action_ids   = (const int*)  d_in[1];
  const int*   street_ids   = (const int*)  d_in[2];
  const float* bet_fraction = (const float*)d_in[3];
  const int*   valid_mask   = (const int*)  d_in[4];
  const float* E_actor      = (const float*)d_in[5];
  const float* E_action     = (const float*)d_in[6];
  const float* E_street     = (const float*)d_in[7];
  const float* W_proj       = (const float*)d_in[8];
  const float* b_proj       = (const float*)d_in[9];
  const float* W_ih         = (const float*)d_in[10];
  const float* W_hh         = (const float*)d_in[11];
  const float* b_ih         = (const float*)d_in[12];
  const float* b_hh         = (const float*)d_in[13];

  const int B = in_sizes[0] / T_STEPS;                   // 8192
  const int rowsPerBlock = WAVES_PER_BLK * ROWS_PER_WAVE; // 32
  dim3 grid((B + rowsPerBlock - 1) / rowsPerBlock);       // 256 blocks
  dim3 block(WAVES_PER_BLK * 32);                         // 64 threads (2 waves)

  gru_hand_history_kernel<<<grid, block, 0, stream>>>(
      actor_ids, action_ids, street_ids, bet_fraction, valid_mask,
      E_actor, E_action, E_street, W_proj, b_proj,
      W_ih, W_hh, b_ih, b_hh, (float*)d_out);
}
